// DualNoneLocal_36009005809720
// MI455X (gfx1250) — matrix operand order repaired and checked by reading, not compiled, not run on hardware
//
#include <hip/hip_runtime.h>

// ---------------------------------------------------------------------------
// DualNonLocal attention for MI455X (gfx1250), wave32, all GEMMs on WMMA f16
// (f32 accumulate).  B=2, C=256, H=64, N=4096, INTER=64.
//
// Every matrix operand is pre-swizzled into a WMMA fragment lane
// layout with k = 16*(lane/16) + e, so each fragment is one aligned 32-byte vector load
// (2x global_load_b128 / ds_load_b128) with zero packing VALU.
// ---------------------------------------------------------------------------

#define N_TOK 4096
#define O_DIM 64
#define C_DIM 256
#define MB    16   // query rows per workgroup in attention kernel

typedef __attribute__((ext_vector_type(16))) _Float16 v16h;
typedef __attribute__((ext_vector_type(8)))  _Float16 v8h;
typedef __attribute__((ext_vector_type(8)))  float    v8f;

#define SHUF16(lo, hi) __builtin_shufflevector(lo, hi, 0,1,2,3,4,5,6,7,8,9,10,11,12,13,14,15)
#define WMMA_F16(a, b, c) __builtin_amdgcn_wmma_f32_16x16x32_f16(false, a, false, b, (short)0, c, false, false)

// fragment sizes (halfs)
#define WFRAG_ELEMS 16384            // per weight matrix: A-fragments
#define SRCSWZ_PER_B ((size_t)256 * 8 * 32 * 16)   // fg/bg B-fragments per batch

// ---------------------------------------------------------------------------
// Kernel 1: pack 7 weight matrices into WMMA A-fragment layout (f16).
//  p in 0..5: w[64][256]  (M=o tiles: ot 0..3, K=c groups: cg 0..7)
//  p == 6  : w_out[256][64] (M=c tiles: ot 0..15, K=o groups: cg 0..1)
// layout: [ot][cg][lane][e]
// ---------------------------------------------------------------------------
__global__ void __launch_bounds__(256)
wswz_kernel(const float* __restrict__ w0, const float* __restrict__ w1,
            const float* __restrict__ w2, const float* __restrict__ w3,
            const float* __restrict__ w4, const float* __restrict__ w5,
            const float* __restrict__ w6, _Float16* __restrict__ wswzAll)
{
    const int p = blockIdx.x;
    const float* w;
    switch (p) {
        case 0: w = w0; break; case 1: w = w1; break; case 2: w = w2; break;
        case 3: w = w3; break; case 4: w = w4; break; case 5: w = w5; break;
        default: w = w6; break;
    }
    _Float16* dst = wswzAll + (size_t)p * WFRAG_ELEMS;
    for (int idx = threadIdx.x; idx < WFRAG_ELEMS; idx += 256) {
        const int e    = idx & 15;
        const int lane = (idx >> 4) & 31;
        int ot, cg;
        if (p < 6) { ot = idx >> 12; cg = (idx >> 9) & 7; }
        else       { ot = idx >> 10; cg = (idx >> 9) & 1; }
        const int r   = lane & 15;
        const int kbA = (lane & 16) ? 16 : 0;
        const int k   = kbA + e;   // A-layout K offset
        float val;
        if (p < 6) val = w[(ot * 16 + r) * C_DIM + cg * 32 + k];   // (M=o, K=c)
        else       val = w[(ot * 16 + r) * O_DIM + cg * 32 + k];   // (M=c, K=o)
        dst[idx] = (_Float16)val;
    }
}

// ---------------------------------------------------------------------------
// Kernel 2: masked fg/bg tokens -> f16 B-fragment swizzle (K = channel c).
// layout: [b][nt(256)][cg(8)][lane(32)][e(16)], element (c, n):
//   cg=c>>5, k=c&31, e=k&15, lane=(n&15)+((k&16)?16:0), nt=n>>4
// one thread = one (b, c, 16-token tile)
// ---------------------------------------------------------------------------
__global__ void __launch_bounds__(256)
cvt_kernel(const float* __restrict__ x, const float* __restrict__ mask,
           _Float16* __restrict__ fgswz, _Float16* __restrict__ bgswz)
{
    const int id = blockIdx.x * 256 + threadIdx.x;   // 2*256*256 threads
    const int nt = id & 255;
    const int c  = (id >> 8) & 255;
    const int b  = id >> 16;

    const float* xr = x + ((size_t)b * C_DIM + c) * N_TOK + nt * 16;
    const float* mr = mask + (size_t)b * N_TOK + nt * 16;

    const int cg = c >> 5, k = c & 31;
    const int e = k & 15, lnAdd = (k & 16) ? 16 : 0;
    const size_t base = (((size_t)b * 256 + nt) * 8 + cg) * 32 * 16;

#pragma unroll
    for (int j = 0; j < 16; ++j) {
        const float xv = xr[j], mv = mr[j];
        const size_t a = base + (size_t)(j + lnAdd) * 16 + e;
        fgswz[a] = (_Float16)(xv * mv);
        bgswz[a] = (_Float16)(xv * (1.0f - mv));
    }
}

// ---------------------------------------------------------------------------
// Kernel 3: six projections via WMMA.  out(o,n) = sum_c w[o,c]*src(c,n)
// M=64 (4 ot tiles), K=256 (8 cg), N = token tiles.
// Output layouts: 0 = Q plain [b][o][n], 1 = K B-frag (phase-1), 2 = V B-frag
// (phase-3).  p = blockIdx.z selects weight/src/output.
// ---------------------------------------------------------------------------
__global__ void __launch_bounds__(256)
proj_wmma(const _Float16* __restrict__ fgswz, const _Float16* __restrict__ bgswz,
          const _Float16* __restrict__ wswzAll,
          _Float16* __restrict__ oQs, _Float16* __restrict__ oKs, _Float16* __restrict__ oVs,
          _Float16* __restrict__ oQc, _Float16* __restrict__ oKc, _Float16* __restrict__ oVc)
{
    const int tid  = threadIdx.x;
    const int lane = tid & 31;
    const int wave = tid >> 5;
    const int b    = blockIdx.y;
    const int p    = blockIdx.z;
    const int nt   = blockIdx.x * 2 + (wave >> 2);   // token tile
    const int ot   = wave & 3;                       // o tile

    const _Float16* src = (p < 4) ? fgswz : bgswz;   // Qs,Ks,Vs,Qc from fg
    int layout; _Float16* out;
    switch (p) {
        case 0: layout = 0; out = oQs; break;
        case 1: layout = 1; out = oKs; break;
        case 2: layout = 2; out = oVs; break;
        case 3: layout = 0; out = oQc; break;
        case 4: layout = 1; out = oKc; break;
        default: layout = 2; out = oVc; break;
    }

    const _Float16* wf = wswzAll + (size_t)p * WFRAG_ELEMS + (size_t)lane * 16;
    const _Float16* sf = src + ((size_t)b * 256 + nt) * 8 * 512 + (size_t)lane * 16;

    v8f cf = {};
#pragma unroll
    for (int cg = 0; cg < 8; ++cg) {
        v16h a  = *(const v16h*)(wf + (size_t)(ot * 8 + cg) * 512);
        v16h bf = *(const v16h*)(sf + (size_t)cg * 512);
        cf = WMMA_F16(a, bf, cf);
    }

    const int colB = lane & 15;
    const int rbC  = (lane < 16) ? 0 : 8;
    const int n    = nt * 16 + colB;
#pragma unroll
    for (int v = 0; v < 8; ++v) {
        const int o = ot * 16 + rbC + v;
        const _Float16 hv = (_Float16)cf[v];
        if (layout == 0) {
            out[((size_t)b * O_DIM + o) * N_TOK + n] = hv;
        } else if (layout == 1) {
            const int g = o >> 5, k = o & 31;
            out[((((size_t)b * 256 + nt) * 2 + g) * 32 + (colB + ((k & 16) ? 16 : 0))) * 16 + (k & 15)] = hv;
        } else {
            const int k2 = n & 31;
            out[((((size_t)b * 128 + (nt >> 1)) * 4 + ot) * 32 + ((o & 15) + ((k2 & 16) ? 16 : 0))) * 16 + colB] = hv;
        }
    }
}

// ---------------------------------------------------------------------------
// Kernel 4: attention.  One workgroup = one (batch, 16-query row block).
// Phase 4 emits feat as f16 B-fragments (K = o) for the output projection:
// featswz layout [b][nt(256)][cg(2)][lane(32)][e(16)]
// ---------------------------------------------------------------------------
__global__ void __launch_bounds__(256)
attn_kernel(const _Float16* __restrict__ Qbuf, const _Float16* __restrict__ Kswz,
            const _Float16* __restrict__ Vswz, float* __restrict__ scoreOut,
            _Float16* __restrict__ featSwz)
{
    extern __shared__ char smem[];
    _Float16* s_p    = (_Float16*)smem;                                  // 16*4096 f16 (128KB)
    _Float16* s_q    = (_Float16*)(smem + (size_t)MB * N_TOK * 2);       // 16*64 f16
    float*    s_feat = (float*)   (smem + (size_t)MB * N_TOK * 2 + MB * O_DIM * 2); // 16*64 f32

    const int tid  = threadIdx.x;
    const int lane = tid & 31;
    const int wave = tid >> 5;              // 8 waves
    const int b    = blockIdx.y;
    const int m0   = blockIdx.x * MB;

    for (int i = tid; i < MB * O_DIM; i += 256) {
        const int m = i >> 6, o = i & 63;
        s_q[i] = Qbuf[((size_t)b * O_DIM + o) * N_TOK + m0 + m];
    }
    for (int i = tid; i < MB * O_DIM; i += 256) s_feat[i] = 0.0f;
    __syncthreads();

    const int r    = lane & 15;
    const int kbA  = (lane < 16) ? 0 : 16;
    const int colB = lane & 15;
    const int rbC  = (lane < 16) ? 0 : 8;

    const _Float16* qrow = s_q + r * O_DIM;
    v16h a0 = *(const v16h*)(qrow + kbA);
    v16h a1 = *(const v16h*)(qrow + 32 + kbA);

    // ---- Phase 1: logits = Q^T K ----
    const _Float16* Kb = Kswz + (size_t)b * 256 * 2 * 32 * 16;
    for (int nt = wave; nt < N_TOK / 16; nt += 8) {
        const _Float16* kt = Kb + ((size_t)nt * 2) * 512 + lane * 16;
        __builtin_prefetch(kt + 8 * 2 * 512, 0, 1);
        v16h b0 = *(const v16h*)(kt);
        v16h b1 = *(const v16h*)(kt + 512);
        v8f c = {};
        c = WMMA_F16(a0, b0, c);
        c = WMMA_F16(a1, b1, c);
#pragma unroll
        for (int v = 0; v < 8; ++v)
            s_p[(rbC + v) * N_TOK + nt * 16 + colB] = (_Float16)c[v];
    }
    __syncthreads();

    // ---- Phase 2: softmax (wave per row), v8h vectorized ----
    for (int rr = wave; rr < MB; rr += 8) {
        _Float16* rowp = s_p + rr * N_TOK;

        float mx = -3.0e38f;
        for (int n = lane * 8; n < N_TOK; n += 32 * 8) {
            v8h v = *(const v8h*)(rowp + n);
#pragma unroll
            for (int e = 0; e < 8; ++e) mx = fmaxf(mx, (float)v[e]);
        }
#pragma unroll
        for (int off = 16; off >= 1; off >>= 1)
            mx = fmaxf(mx, __shfl_xor(mx, off, 32));

        float sum = 0.0f;
        for (int n = lane * 8; n < N_TOK; n += 32 * 8) {
            v8h v = *(const v8h*)(rowp + n);
#pragma unroll
            for (int e = 0; e < 8; ++e) sum += __expf((float)v[e] - mx);
        }
#pragma unroll
        for (int off = 16; off >= 1; off >>= 1)
            sum += __shfl_xor(sum, off, 32);
        const float inv = 1.0f / sum;

        float* so = scoreOut + ((size_t)b * N_TOK + (m0 + rr)) * N_TOK;
        for (int n = lane * 8; n < N_TOK; n += 32 * 8) {
            v8h v = *(const v8h*)(rowp + n);
            v8f pf; v8h ph;
#pragma unroll
            for (int e = 0; e < 8; ++e) {
                const float p = __expf((float)v[e] - mx) * inv;
                pf[e] = p; ph[e] = (_Float16)p;
            }
            *(v8f*)(so + n)   = pf;   // 2x global_store_b128
            *(v8h*)(rowp + n) = ph;   // ds_store_b128 in-place
        }
    }
    __syncthreads();

    // ---- Phase 3: feat[m,o] = sum_n P[m,n] * V[o,n] ----
    const _Float16* Vb = Vswz + (size_t)b * 128 * 4 * 32 * 16;
    const int ot = wave & 3;
    const int nh = wave >> 2;
    const _Float16* prow = s_p + r * N_TOK;
    v8f cf = {};
    for (int n0 = nh * (N_TOK / 2); n0 < nh * (N_TOK / 2) + (N_TOK / 2); n0 += 32) {
        v16h ap = *(const v16h*)(prow + n0 + kbA);
        v16h bv = *(const v16h*)(Vb + (((size_t)(n0 >> 5) * 4 + ot) * 32 + lane) * 16);
        cf = WMMA_F16(ap, bv, cf);
    }
#pragma unroll
    for (int v = 0; v < 8; ++v)
        atomicAdd(&s_feat[(rbC + v) * O_DIM + ot * 16 + colB], cf[v]);   // ds_add_f32
    __syncthreads();

    // ---- Phase 4: feat tile -> f16 B-fragments (K = o) ----
    _Float16* fo = featSwz + ((size_t)b * 256 + (m0 >> 4)) * 2 * 512;
    for (int i = tid; i < MB * O_DIM; i += 256) {
        const int m = i >> 6, o = i & 63;
        const int cg = o >> 5, k = o & 31;
        fo[((size_t)cg * 32 + (m + ((k & 16) ? 16 : 0))) * 16 + (k & 15)] = (_Float16)s_feat[m * O_DIM + o];
    }
}

// ---------------------------------------------------------------------------
// Kernel 5: out[b,c,n] = x + sum_o w_out[c,o]*(featS+featC)[b,o,n] via WMMA.
// M=256 (16 ot tiles), K=64 (2 cg), one workgroup per (token tile, batch).
// Branch feats are summed as f16 fragments (v_pk_add_f16) before the WMMA.
// ---------------------------------------------------------------------------
__global__ void __launch_bounds__(256)
outproj_wmma(const float* __restrict__ x, const _Float16* __restrict__ featS,
             const _Float16* __restrict__ featC, const _Float16* __restrict__ woutswz,
             float* __restrict__ out)
{
    const int tid  = threadIdx.x;
    const int lane = tid & 31;
    const int wave = tid >> 5;
    const int nt   = blockIdx.x;
    const int b    = blockIdx.y;

    const int colB = lane & 15;
    const int rbC  = (lane < 16) ? 0 : 8;
    const int n    = nt * 16 + colB;

    const _Float16* fS = featS + ((size_t)b * 256 + nt) * 2 * 512 + (size_t)lane * 16;
    const _Float16* fC = featC + ((size_t)b * 256 + nt) * 2 * 512 + (size_t)lane * 16;
    const float* xb = x   + (size_t)b * C_DIM * N_TOK;
    float*       ob = out + (size_t)b * C_DIM * N_TOK;

    // preload + sum the two branch fragments once (shared across all ot tiles)
    v16h bsum0 = *(const v16h*)(fS)       + *(const v16h*)(fC);        // cg=0
    v16h bsum1 = *(const v16h*)(fS + 512) + *(const v16h*)(fC + 512);  // cg=1

    for (int ot = wave; ot < 16; ot += 8) {
        const _Float16* wf = woutswz + (size_t)(ot * 2) * 512 + (size_t)lane * 16;
        v8f cf = {};
        cf = WMMA_F16(*(const v16h*)(wf),       bsum0, cf);
        cf = WMMA_F16(*(const v16h*)(wf + 512), bsum1, cf);
#pragma unroll
        for (int v = 0; v < 8; ++v) {
            const int c = ot * 16 + rbC + v;
            ob[(size_t)c * N_TOK + n] = xb[(size_t)c * N_TOK + n] + cf[v];
        }
    }
}

// ---------------------------------------------------------------------------
extern "C" void kernel_launch(void* const* d_in, const int* in_sizes, int n_in,
                              void* d_out, int out_size, void* d_ws, size_t ws_size,
                              hipStream_t stream)
{
    (void)in_sizes; (void)n_in; (void)out_size; (void)ws_size;

    const float* x         = (const float*)d_in[0];
    const float* mask      = (const float*)d_in[1];
    const float* w_theta_s = (const float*)d_in[2];
    const float* w_phi_s   = (const float*)d_in[3];
    const float* w_gate_s  = (const float*)d_in[4];
    const float* w_theta_c = (const float*)d_in[5];
    const float* w_phi_c   = (const float*)d_in[6];
    const float* w_gate_c  = (const float*)d_in[7];
    const float* w_out     = (const float*)d_in[8];
    float*       out       = (float*)d_out;

    // ---- workspace layout (halfs), ~16.2 MB total ----
    const size_t QKV = (size_t)2 * O_DIM * N_TOK;            // 524288
    _Float16* ws16 = (_Float16*)d_ws;
    _Float16* Qs = ws16;            _Float16* Ks = ws16 + QKV;     _Float16* Vs = ws16 + 2 * QKV;
    _Float16* Qc = ws16 + 3 * QKV;  _Float16* Kc = ws16 + 4 * QKV; _Float16* Vc = ws16 + 5 * QKV;
    _Float16* fgswz   = ws16 + 6 * QKV;                       // 2*SRCSWZ_PER_B
    _Float16* bgswz   = fgswz + 2 * SRCSWZ_PER_B;
    _Float16* wswzAll = bgswz + 2 * SRCSWZ_PER_B;             // 7*16384
    _Float16* featSsw = wswzAll + 7 * WFRAG_ELEMS;            // 524288
    _Float16* featCsw = featSsw + QKV;

    const dim3 blk(256);

    wswz_kernel<<<7, blk, 0, stream>>>(w_phi_s, w_theta_s, w_gate_s,
                                       w_phi_c, w_theta_c, w_gate_c, w_out, wswzAll);
    cvt_kernel<<<512, blk, 0, stream>>>(x, mask, fgswz, bgswz);

    proj_wmma<<<dim3(128, 2, 6), blk, 0, stream>>>(fgswz, bgswz, wswzAll,
                                                   Qs, Ks, Vs, Qc, Kc, Vc);

    const size_t featElems  = (size_t)2 * C_DIM * N_TOK;          // 2,097,152
    const size_t scoreElems = (size_t)2 * N_TOK * N_TOK;          // 33,554,432
    float* scoreSelf  = out + featElems;
    float* scoreCross = scoreSelf + scoreElems;

    const size_t shBytes = (size_t)MB * N_TOK * 2 + (size_t)MB * O_DIM * 2 + (size_t)MB * O_DIM * 4;
    const dim3 gA(N_TOK / MB, 2);
    attn_kernel<<<gA, blk, shBytes, stream>>>(Qs, Ks, Vs, scoreSelf,  featSsw);
    attn_kernel<<<gA, blk, shBytes, stream>>>(Qc, Kc, Vc, scoreCross, featCsw);

    outproj_wmma<<<dim3(256, 2), blk, 0, stream>>>(x, featSsw, featCsw,
                                                   wswzAll + 6 * WFRAG_ELEMS, out);
}
